// ScaledDotProductAttention_27676769255543
// MI455X (gfx1250) — matrix operand order repaired and checked by reading, not compiled, not run on hardware
//
#include <hip/hip_runtime.h>
#include <hip/hip_bf16.h>

typedef __attribute__((ext_vector_type(16))) _Float16 v16h;
typedef __attribute__((ext_vector_type(8)))  _Float16 v8h;
typedef __attribute__((ext_vector_type(4)))  _Float16 v4h;
typedef __attribute__((ext_vector_type(8)))  float    v8f;
typedef __attribute__((ext_vector_type(4)))  float    v4f;

#define S_LEN   2048
#define D_HEAD  64
#define NWAVES  8
#define BM      (NWAVES * 16) // 128 query rows per block
#define BN      64            // KV chunk
#define VS      72            // padded t-stride for transposed V in LDS (144B, 16B-mult)
#define NBH     32            // B*H
#define TEMP_INV 0.125f       // 1/sqrt(64)
#define EPS_MAG  1e-12f

static __device__ __forceinline__ v8f wmma16(v16h a, v16h b, v8f c) {
  // v_wmma_f32_16x16x32_f16, fp32 accumulate
  return __builtin_amdgcn_wmma_f32_16x16x32_f16(false, a, false, b, (short)0, c, false, false);
}

static __device__ __forceinline__ v16h ld16(const _Float16* p) {
  v16h o;
#pragma unroll
  for (int j = 0; j < 16; ++j) o[j] = p[j];
  return o;
}

// Build a 16-bit A-fragment from two contiguous 8-half groups (16B-aligned).
static __device__ __forceinline__ v16h lda(const _Float16* p) {
  const v8h lo = *(const v8h*)p;
  const v8h hi = *(const v8h*)(p + 8);
  v16h o;
#pragma unroll
  for (int j = 0; j < 8; ++j) { o[j] = lo[j]; o[8 + j] = hi[j]; }
  return o;
}

__global__ __launch_bounds__(NWAVES * 32)
void cvattn_kernel(const float* __restrict__ qr, const float* __restrict__ qi,
                   const float* __restrict__ kr, const float* __restrict__ ki,
                   const float* __restrict__ vr, const float* __restrict__ vi,
                   float* __restrict__ out)
{
  __shared__ _Float16 sKr[BN * D_HEAD];          // [t][d] f16           8 KB
  __shared__ _Float16 sKi[BN * D_HEAD];          //                      8 KB
  __shared__ _Float16 sVr[D_HEAD * VS];          // [d][t] f16 (transp)  9 KB
  __shared__ _Float16 sVi[D_HEAD * VS];          //                      9 KB
  __shared__ _Float16 sP[NWAVES][16 * BN];       // per-wave P scratch  16 KB

  const int tid  = threadIdx.x;
  const int wave = tid >> 5;
  const int lane = tid & 31;
  const int hf   = lane >> 4;      // half-wave: 0 or 1
  const int r    = lane & 15;

  const int bh = blockIdx.y;
  const int m0 = blockIdx.x * BM + wave * 16;    // wave's first query row
  const size_t base = (size_t)bh * S_LEN * D_HEAD;

  const float* Qr = qr + base; const float* Qi = qi + base;
  const float* Kr = kr + base; const float* Ki = ki + base;
  const float* Vr = vr + base; const float* Vi = vi + base;

  // ---- Q A-fragments (f32 -> f16), plus negated imag for the real-part GEMM.
  // A layout used here (16x32): lanes 0-15: row=r, K = kt*32 + 0..15 (contiguous);
  //                           lanes 16-31: same rows, K shifted by +16.
  v16h aQr[2], aQi[2], aQin[2];
  {
    const float* rq = Qr + (size_t)(m0 + r) * D_HEAD;
    const float* ri = Qi + (size_t)(m0 + r) * D_HEAD;
#pragma unroll
    for (int kt = 0; kt < 2; ++kt) {
      const int c0 = kt * 32 + hf * 16;
#pragma unroll
      for (int j = 0; j < 8; ++j) {
        float a0 = rq[c0 + j],      a1 = rq[c0 + 8 + j];
        float b0 = ri[c0 + j],      b1 = ri[c0 + 8 + j];
        aQr[kt][j]  = (_Float16)a0;    aQr[kt][8 + j]  = (_Float16)a1;
        aQi[kt][j]  = (_Float16)b0;    aQi[kt][8 + j]  = (_Float16)b1;
        aQin[kt][j] = (_Float16)(-b0); aQin[kt][8 + j] = (_Float16)(-b1);
      }
    }
  }

  // Output accumulators (C layout), online-softmax row stats.
  v8f accR[4], accI[4];
  float m_i[8], l_i[8];
#pragma unroll
  for (int n = 0; n < 4; ++n)
#pragma unroll
    for (int e = 0; e < 8; ++e) { accR[n][e] = 0.f; accI[n][e] = 0.f; }
#pragma unroll
  for (int j = 0; j < 8; ++j) { m_i[j] = -1e30f; l_i[j] = 0.f; }

  _Float16* myP = &sP[wave][0];

  for (int nb = 0; nb < S_LEN / BN; ++nb) {
    const int t0 = nb * BN;

    __syncthreads();   // previous chunk's K/V LDS reads complete before restage
    // ---- stage K (row-major) and V (transposed) as f16; 4 float4s/thread/array
#pragma unroll
    for (int it = 0; it < (BN * D_HEAD / 4) / (NWAVES * 32); ++it) {   // 4 iters
      const int idx = tid + it * (NWAVES * 32);
      const int e = idx * 4;
      const int t = e >> 6;
      const int d = e & 63;
      const v4f fr = *(const v4f*)(Kr + (size_t)(t0 + t) * D_HEAD + d);
      const v4f fi = *(const v4f*)(Ki + (size_t)(t0 + t) * D_HEAD + d);
      v4h hr, hi;
#pragma unroll
      for (int j = 0; j < 4; ++j) { hr[j] = (_Float16)fr[j]; hi[j] = (_Float16)fi[j]; }
      *(v4h*)(sKr + t * D_HEAD + d) = hr;
      *(v4h*)(sKi + t * D_HEAD + d) = hi;
      const v4f gr = *(const v4f*)(Vr + (size_t)(t0 + t) * D_HEAD + d);
      const v4f gi = *(const v4f*)(Vi + (size_t)(t0 + t) * D_HEAD + d);
#pragma unroll
      for (int j = 0; j < 4; ++j) {
        sVr[(d + j) * VS + t] = (_Float16)gr[j];
        sVi[(d + j) * VS + t] = (_Float16)gi[j];
      }
    }
    __syncthreads();

    // ---- scores: Sr = QrKr^T - QiKi^T ; Si = QrKi^T + QiKr^T  (16x64 per wave)
    v8f sr[4], si[4];
#pragma unroll
    for (int n = 0; n < 4; ++n)
#pragma unroll
      for (int e = 0; e < 8; ++e) { sr[n][e] = 0.f; si[n][e] = 0.f; }
#pragma unroll
    for (int n = 0; n < 4; ++n) {
#pragma unroll
      for (int kt = 0; kt < 2; ++kt) {
        // B frag (K^T tile): lane = col t = n*16+r, K(=d) = kt*32 + hf*16 .. +15
        const v16h bkr = ld16(sKr + (n * 16 + r) * D_HEAD + kt * 32 + hf * 16);
        const v16h bki = ld16(sKi + (n * 16 + r) * D_HEAD + kt * 32 + hf * 16);
        sr[n] = wmma16(aQr[kt],  bkr, sr[n]);
        sr[n] = wmma16(aQin[kt], bki, sr[n]);   // -Qi*Ki
        si[n] = wmma16(aQr[kt],  bki, si[n]);
        si[n] = wmma16(aQi[kt],  bkr, si[n]);
      }
    }

    // ---- online magnitude-softmax with phase; Pr -> per-wave LDS, Pi kept in regs
#pragma unroll
    for (int j = 0; j < 8; ++j) {
      float mg[4], ms[4];
#pragma unroll
      for (int n = 0; n < 4; ++n) {
        const float a = sr[n][j], b = si[n][j];
        mg[n] = sqrtf(a * a + b * b);           // raw |z|
        ms[n] = mg[n] * TEMP_INV;               // scaled magnitude for softmax
      }
      float mx = fmaxf(fmaxf(ms[0], ms[1]), fmaxf(ms[2], ms[3]));
#pragma unroll
      for (int o = 1; o < 16; o <<= 1) mx = fmaxf(mx, __shfl_xor(mx, o, 32));
      const float mnew  = fmaxf(m_i[j], mx);
      const float scale = __expf(m_i[j] - mnew);
      m_i[j] = mnew;
      float rs = 0.f;
#pragma unroll
      for (int n = 0; n < 4; ++n) {
        const float p    = __expf(ms[n] - mnew);
        const float coef = p / fmaxf(mg[n], EPS_MAG);  // p * phase / |z|
        sr[n][j] *= coef;                               // -> P_real
        si[n][j] *= coef;                               // -> P_imag
        rs += p;
      }
#pragma unroll
      for (int o = 1; o < 16; o <<= 1) rs += __shfl_xor(rs, o, 32);
      l_i[j] = l_i[j] * scale + rs;
#pragma unroll
      for (int n = 0; n < 4; ++n) { accR[n][j] *= scale; accI[n][j] *= scale; }
      const int rowoff = (j + hf * 8) * BN;             // C layout: row = j + hf*8
#pragma unroll
      for (int n = 0; n < 4; ++n)
        myP[rowoff + n * 16 + r] = (_Float16)sr[n][j];  // Pr only
    }

    // ---- sweep 1: Pr contributions.  Same-wave DS ops are in order: no barrier.
    v16h aP[2];
#pragma unroll
    for (int kt = 0; kt < 2; ++kt)
      aP[kt] = lda(myP + r * BN + kt * 32 + hf * 16);
#pragma unroll
    for (int n = 0; n < 4; ++n) {
#pragma unroll
      for (int kt = 0; kt < 2; ++kt) {
        // B frag (V tile): lane = col d = n*16+r, K(=t) = kt*32 + hf*16 .. +15
        const v16h bvr = ld16(sVr + (n * 16 + r) * VS + kt * 32 + hf * 16);
        const v16h bvi = ld16(sVi + (n * 16 + r) * VS + kt * 32 + hf * 16);
        accR[n] = wmma16(aP[kt], bvr, accR[n]);   // +Pr*Vr
        accI[n] = wmma16(aP[kt], bvi, accI[n]);   // +Pr*Vi
      }
    }

    // ---- store Pi into the same buffer (in-order DS: prior loads already read Pr)
#pragma unroll
    for (int j = 0; j < 8; ++j) {
      const int rowoff = (j + hf * 8) * BN;
#pragma unroll
      for (int n = 0; n < 4; ++n)
        myP[rowoff + n * 16 + r] = (_Float16)si[n][j];
    }

    // ---- sweep 2: Pi contributions
    v16h aPn[2];
#pragma unroll
    for (int kt = 0; kt < 2; ++kt) {
      aP[kt] = lda(myP + r * BN + kt * 32 + hf * 16);
#pragma unroll
      for (int j = 0; j < 16; ++j) aPn[kt][j] = -aP[kt][j];
    }
#pragma unroll
    for (int n = 0; n < 4; ++n) {
#pragma unroll
      for (int kt = 0; kt < 2; ++kt) {
        const v16h bvr = ld16(sVr + (n * 16 + r) * VS + kt * 32 + hf * 16);
        const v16h bvi = ld16(sVi + (n * 16 + r) * VS + kt * 32 + hf * 16);
        accR[n] = wmma16(aPn[kt], bvi, accR[n]);  // -Pi*Vi
        accI[n] = wmma16(aP[kt],  bvr, accI[n]);  // +Pi*Vr
      }
    }
  }

  // ---- normalize and write out: [2, B, H, S, D]
  const size_t imag_off = (size_t)NBH * S_LEN * D_HEAD;
  float* Or = out + base;
  float* Oi = out + imag_off + base;
#pragma unroll
  for (int j = 0; j < 8; ++j) {
    const float invl = 1.0f / l_i[j];
    const int row = m0 + j + hf * 8;
#pragma unroll
    for (int n = 0; n < 4; ++n) {
      Or[(size_t)row * D_HEAD + n * 16 + r] = accR[n][j] * invl;
      Oi[(size_t)row * D_HEAD + n * 16 + r] = accI[n][j] * invl;
    }
  }
}

extern "C" void kernel_launch(void* const* d_in, const int* in_sizes, int n_in,
                              void* d_out, int out_size, void* d_ws, size_t ws_size,
                              hipStream_t stream) {
  (void)in_sizes; (void)n_in; (void)out_size; (void)d_ws; (void)ws_size;
  const float* qr = (const float*)d_in[0];
  const float* qi = (const float*)d_in[1];
  const float* kr = (const float*)d_in[2];
  const float* ki = (const float*)d_in[3];
  const float* vr = (const float*)d_in[4];
  const float* vi = (const float*)d_in[5];
  dim3 grid(S_LEN / BM, NBH);           // 16 x 32; x-fastest keeps one head's K/V hot in L2
  cvattn_kernel<<<grid, NWAVES * 32, 0, stream>>>(qr, qi, kr, ki, vr, vi, (float*)d_out);
}
